// OFT_30666066493909
// MI455X (gfx1250) — compile-verified, matrix operand order checked by reading, not run on hardware
//
#include <hip/hip_runtime.h>
#include <hip/hip_bf16.h>

typedef __attribute__((ext_vector_type(16))) __bf16 v16bf;
typedef __attribute__((ext_vector_type(8)))  float  v8f;

// Problem constants (from reference setup_inputs)
#define C_      256
#define HF      96
#define WF      312
#define DG      160      // depth grid points
#define WG      160      // width grid points
#define DEPTH   159
#define WIDTH   159
#define P_      (DEPTH*WIDTH)   // 25281
#define NCELL   7
#define KDIM    (C_*NCELL)      // 1792
#define KT      (KDIM/32)       // 56
#define NT      (C_/16)         // 16
#define MT      ((P_+15)/16)    // 1581 M-tiles

__device__ __forceinline__ unsigned short f2bf(float f) {
    union { float f; unsigned u; } v; v.f = f;
    unsigned u = v.u;
    u += 0x7FFFu + ((u >> 16) & 1u);       // round-to-nearest-even
    return (unsigned short)(u >> 16);
}

// K-position within a 32-wide K-slice for 16-bit A/B fragments:
// K(l,e) = ((e>>3)<<4) | ((l>>4)<<3) | (e&7)

// ---------------- Kernel 1: row-wise inclusive scan (cumsum over W) ----------
__global__ __launch_bounds__(256) void scan_rows(const float* __restrict__ f,
                                                 float* __restrict__ tmp) {
    int row  = blockIdx.x * 8 + (threadIdx.x >> 5);   // C*HF rows
    int lane = threadIdx.x & 31;
    if (row >= C_ * HF) return;
    const float* src = f   + (size_t)row * WF;
    float*       dst = tmp + (size_t)row * WF;
    float carry = 0.f;
    for (int base = 0; base < WF; base += 32) {
        int idx = base + lane;
        float v = (idx < WF) ? src[idx] : 0.f;
        #pragma unroll
        for (int off = 1; off < 32; off <<= 1) {
            float t = __shfl_up(v, off, 32);
            if (lane >= off) v += t;
        }
        v += carry;
        if (idx < WF) dst[idx] = v;
        carry = __shfl(v, 31, 32);
    }
}

// ------- Kernel 2: column cumsum + transpose to channel-last (H, W, C) -------
__global__ __launch_bounds__(256) void scan_cols(const float* __restrict__ tmp,
                                                 float* __restrict__ iit) {
    int t = blockIdx.x * 256 + threadIdx.x;           // C*WF threads
    if (t >= C_ * WF) return;
    int c = t / WF, w = t % WF;                       // reads coalesced over w
    float run = 0.f;
    for (int h = 0; h < HF; ++h) {
        run += tmp[(size_t)c * HF * WF + (size_t)h * WF + w];
        iit[((size_t)h * WF + w) * C_ + c] = run;
    }
}

// ------ Kernel 3: pack W_lin (C x K, f32) into bf16 WMMA-B fragment order ----
__global__ __launch_bounds__(256) void pack_w(const float* __restrict__ wl,
                                              unsigned short* __restrict__ wp) {
    int t = blockIdx.x * 256 + threadIdx.x;           // NT*KT*32 threads
    if (t >= NT * KT * 32) return;
    int l   = t & 31;
    int kt  = (t >> 5) % KT;
    int nt  = t / (KT * 32);
    int co  = nt * 16 + (l & 15);
    int lhi = l >> 4;
    #pragma unroll
    for (int e = 0; e < 16; ++e) {
        int kk = ((e >> 3) << 4) | (lhi << 3) | (e & 7);
        wp[(size_t)t * 16 + e] = f2bf(wl[(size_t)co * KDIM + kt * 32 + kk]);
    }
}

// ---------------- Kernel 4: fused sampling + bf16 WMMA GEMM ------------------
__global__ __launch_bounds__(256) void oft_fused(const float* __restrict__ calib,
                                                 const float* __restrict__ grid,
                                                 const float* __restrict__ ycorn,
                                                 const float* __restrict__ iit,
                                                 const unsigned short* __restrict__ wpack,
                                                 const float* __restrict__ b_lin,
                                                 float* __restrict__ out) {
    __shared__ __align__(32) unsigned short ldsA[KT * 32 * 16];   // 56 KB: A tile, fragment order
    __shared__ float ldsD[8 * 256];                               // 8 KB: per-wave D staging

    const int tid  = threadIdx.x;
    const int wave = tid >> 5;
    const int lane = tid & 31;
    const int ptile = blockIdx.x;

    float cb[12];
    #pragma unroll
    for (int i = 0; i < 12; ++i) cb[i] = calib[i];

    // ---------------- Phase 1: build vox A-tile (16 positions x 1792 K) -----
    // 16 pos * 7 cells = 112 entries; 8 waves * 14 entries each.
    for (int i = 0; i < 14; ++i) {
        int e  = wave * 14 + i;
        int pl = e / NCELL;           // local position 0..15
        int n  = e % NCELL;           // cell 0..6
        int p  = ptile * 16 + pl;

        float wts[16];
        int   offs[16];
        if (p < P_) {
            int d = p / WIDTH, w = p % WIDTH;
            // grid/y_corners inputs: grid (DG,WG,3) xz-plane, y_corners (8,3)
            float X0 = grid[((size_t)d * WG + w) * 3 + 0];
            float Z0 = grid[((size_t)d * WG + w) * 3 + 2];
            float X1 = grid[((size_t)d * WG + (w + 1)) * 3 + 0];
            float Z1 = grid[((size_t)(d + 1) * WG + w) * 3 + 2];
            float Yn  = ycorn[(size_t)n * 3 + 1];
            float Yn1 = ycorn[(size_t)(n + 1) * 3 + 1];

            float pts[4][3] = { {X0, Yn,  Z0},     // nc[n,   d,   w]
                                {X0, Yn,  Z1},     // nc[n,   d+1, w]
                                {X1, Yn1, Z1},     // nc[n+1, d+1, w+1]
                                {X1, Yn1, Z0} };   // nc[n+1, d,   w+1]
            float ncx[4], ncy[4];
            #pragma unroll
            for (int q = 0; q < 4; ++q) {
                float X = pts[q][0], Y = pts[q][1], Z = pts[q][2];
                float h0 = cb[0] * X + cb[1] * Y + cb[2]  * Z + cb[3];
                float h1 = cb[4] * X + cb[5] * Y + cb[6]  * Z + cb[7];
                float h2 = cb[8] * X + cb[9] * Y + cb[10] * Z + cb[11];
                float inv = 1.0f / h2;
                float u = h0 * inv, v = h1 * inv;
                ncx[q] = fminf(fmaxf(2.f * u * 0.125f / (float)WF - 1.f, -1.f), 1.f);
                ncy[q] = fminf(fmaxf(2.f * v * 0.125f / (float)HF - 1.f, -1.f), 1.f);
            }
            float x0 = fminf(ncx[0], ncx[1]), y0 = fminf(ncy[0], ncy[1]);
            float x1 = fmaxf(ncx[2], ncx[3]), y1 = fmaxf(ncy[2], ncy[3]);
            float rawarea = (x1 - x0) * (y1 - y0) * (float)(HF * WF) * 0.25f;
            float scale = (rawarea > 0.f) ? (1.0f / (rawarea + 1e-6f)) : 0.0f;

            float sxs[4] = {x0, x1, x1, x0};
            float sys[4] = {y0, y1, y0, y1};
            float sgn[4] = {1.f, 1.f, -1.f, -1.f};
            #pragma unroll
            for (int s = 0; s < 4; ++s) {
                float px = (sxs[s] + 1.f) * 0.5f * (float)WF - 0.5f;
                float py = (sys[s] + 1.f) * 0.5f * (float)HF - 0.5f;
                float fx = floorf(px), fy = floorf(py);
                float wx = px - fx, wy = py - fy;
                int ix = (int)fx, iy = (int)fy;
                float bw[4] = {(1.f - wy) * (1.f - wx), (1.f - wy) * wx,
                               wy * (1.f - wx),         wy * wx};
                #pragma unroll
                for (int tt = 0; tt < 4; ++tt) {
                    int xx = ix + (tt & 1);
                    int yy = iy + (tt >> 1);
                    bool valid = (xx >= 0) & (xx < WF) & (yy >= 0) & (yy < HF);
                    int cxx = min(max(xx, 0), WF - 1);
                    int cyy = min(max(yy, 0), HF - 1);
                    offs[s * 4 + tt] = cyy * WF + cxx;
                    wts[s * 4 + tt]  = valid ? sgn[s] * bw[tt] * scale : 0.f;
                }
            }
        } else {
            #pragma unroll
            for (int tt = 0; tt < 16; ++tt) { wts[tt] = 0.f; offs[tt] = 0; }
        }

        // gather 16 taps x 8 channels per lane (channel-last -> coalesced)
        const int cbase = lane * 8;
        float4 a0 = {0.f, 0.f, 0.f, 0.f}, a1 = {0.f, 0.f, 0.f, 0.f};
        #pragma unroll
        for (int tt = 0; tt < 16; ++tt) {
            const float* src = iit + (size_t)offs[tt] * C_ + cbase;
            float4 v0 = *(const float4*)(src);
            float4 v1 = *(const float4*)(src + 4);
            float wt = wts[tt];
            a0.x += wt * v0.x; a0.y += wt * v0.y; a0.z += wt * v0.z; a0.w += wt * v0.w;
            a1.x += wt * v1.x; a1.y += wt * v1.y; a1.z += wt * v1.z; a1.w += wt * v1.w;
        }
        float vals[8] = {a0.x, a0.y, a0.z, a0.w, a1.x, a1.y, a1.z, a1.w};
        #pragma unroll
        for (int j = 0; j < 8; ++j) {
            int c  = cbase + j;
            int k  = c * NCELL + n;           // K ordering: k = c*7 + n
            int kt = k >> 5, kk = k & 31;
            int lrow = ((kk >> 3) & 1) * 16 + pl;
            int ee   = ((kk >> 4) << 3) | (kk & 7);
            ldsA[(kt * 32 + lrow) * 16 + ee] = f2bf(vals[j]);
        }
    }
    __syncthreads();

    // ---------------- Phase 2: WMMA GEMM, 8 waves x 2 N-tiles each ----------
    for (int half = 0; half < 2; ++half) {
        int nt = wave * 2 + half;
        v8f acc = {};
        for (int kt = 0; kt < KT; ++kt) {
            v16bf afrag = *reinterpret_cast<const v16bf*>(&ldsA[(kt * 32 + lane) * 16]);
            v16bf bfrag = *reinterpret_cast<const v16bf*>(
                              wpack + ((size_t)(nt * KT + kt) * 32 + lane) * 16);
            acc = __builtin_amdgcn_wmma_f32_16x16x32_bf16(
                      false, afrag, false, bfrag, (short)0, acc, false, false);
        }
        // stage D tile through LDS -> coalesced bias+relu stores
        float* dst = ldsD + wave * 256;
        int nloc = lane & 15, mhi = (lane >> 4) * 8;
        #pragma unroll
        for (int r = 0; r < 8; ++r) dst[(mhi + r) * 16 + nloc] = acc[r];
        __syncthreads();
        #pragma unroll
        for (int it = 0; it < 8; ++it) {
            int idx  = it * 32 + lane;
            int co_l = idx >> 4;
            int m    = idx & 15;
            int co   = nt * 16 + co_l;
            int p    = ptile * 16 + m;
            float v  = ldsD[wave * 256 + m * 16 + co_l] + b_lin[co];
            v = fmaxf(v, 0.f);
            if (p < P_) out[(size_t)co * P_ + p] = v;
        }
        __syncthreads();
    }
}

extern "C" void kernel_launch(void* const* d_in, const int* in_sizes, int n_in,
                              void* d_out, int out_size, void* d_ws, size_t ws_size,
                              hipStream_t stream) {
    const float* features = (const float*)d_in[0];   // (1,256,96,312)
    const float* calib    = (const float*)d_in[1];   // (1,3,4)
    const float* grid     = (const float*)d_in[2];   // (1,160,160,3)
    const float* ycorn    = (const float*)d_in[3];   // (8,1,1,3)
    const float* w_lin    = (const float*)d_in[4];   // (256,1792)
    const float* b_lin    = (const float*)d_in[5];   // (256,)
    float* out = (float*)d_out;                      // (1,256,159,159)

    char* ws = (char*)d_ws;
    size_t featBytes = (size_t)C_ * HF * WF * sizeof(float);     // 30.7 MB
    float* tmp = (float*)(ws);                                   // row-scanned
    float* iit = (float*)(ws + featBytes);                       // (H,W,C) integral
    unsigned short* wpack = (unsigned short*)(ws + 2 * featBytes); // 0.9 MB bf16 B-frags

    scan_rows<<<(C_ * HF) / 8, 256, 0, stream>>>(features, tmp);
    scan_cols<<<(C_ * WF + 255) / 256, 256, 0, stream>>>(tmp, iit);
    pack_w<<<(NT * KT * 32 + 255) / 256, 256, 0, stream>>>(w_lin, wpack);
    oft_fused<<<MT, 256, 0, stream>>>(calib, grid, ycorn, iit, wpack, b_lin, out);
}
